// GatedActivation_35021163331880
// MI455X (gfx1250) — matrix operand order repaired and checked by reading, not compiled, not run on hardware
//
#include <hip/hip_runtime.h>
#include <hip/hip_bf16.h>

typedef __attribute__((ext_vector_type(16))) _Float16 v16h;
typedef __attribute__((ext_vector_type(8)))  _Float16 v8h;
typedef __attribute__((ext_vector_type(8)))  float    v8f;

#define CIN   120
#define SPAT  110592          // 48*48*48
#define HD    8               // halo depth  (4 + 4)
#define HH    8               // halo height (4 + 4)
#define HW    20              // halo width  (16 + 4)
#define POSH  40              // padded halfs per halo position (32 ch + 8 pad) -> bank-conflict-free
#define NPOS  (HD*HH*HW)      // 1280 halo positions
#define LDSH  (NPOS*POSH)     // 51200 halfs = 100 KB

// ---------------------------------------------------------------------------
// Pack w_gate (24 x 120 x 5x5x5, f32) into per-lane WMMA A-fragment order:
//   A[cc][ko][mt][lane][16 halfs], co padded 24->32, ci padded 120->128.
// CDNA5 16-bit A-matrix 16x32 layout: lanes 0-15 (row M=lane) hold
// K={0..7,16..23}; lanes 16-31 (row M=lane-16) hold K={8..15,24..31}.
// ---------------------------------------------------------------------------
__global__ __launch_bounds__(256)
void GatedAct_packA(const float* __restrict__ wg, _Float16* __restrict__ A)
{
    int t = blockIdx.x * 256 + threadIdx.x;       // 125 blocks * 256 = 32000
    if (t >= 32000) return;
    int lane = t & 31;
    int mt   = (t >> 5) & 1;
    int cc   = (t >> 6) & 3;
    int ko   = t >> 8;                            // 0..124
    int co   = mt * 16 + (lane & 15);

    _Float16 vals[16];
#pragma unroll
    for (int j = 0; j < 16; ++j) {
        int K  = j + (j >= 8 ? 8 : 0) + (lane >= 16 ? 8 : 0);
        int ci = cc * 32 + K;
        float v = 0.0f;
        if (co < 24 && ci < CIN)
            v = wg[((size_t)co * CIN + ci) * 125 + ko];
        vals[j] = (_Float16)v;
    }
    size_t off = ((((size_t)cc * 125 + ko) * 2 + mt) * 32 + lane) * 16;
#pragma unroll
    for (int j = 0; j < 16; ++j) A[off + j] = vals[j];
}

// ---------------------------------------------------------------------------
// Implicit-GEMM 5x5x5 conv (24 gates, padded to 32) via v_wmma_f32_16x16x32_f16,
// fused sigmoid + gated-activation epilogue.
// Grid: 864 blocks (2 batches * 12*12*3 tiles of 4x4x16), 128 threads (4 waves).
// Per wave: 2 M-tiles x 4 N-tiles, accumulators resident in VGPRs.
// ---------------------------------------------------------------------------
__global__ __launch_bounds__(128)
void GatedAct_conv(const float* __restrict__ x,
                   const _Float16* __restrict__ Afrag,
                   float* __restrict__ out)
{
    __shared__ __align__(32) _Float16 lds_x[LDSH];   // 100 KB halo slab (padded)
    float* lds_g = (float*)lds_x;                    // 24*256 f32 = 24 KB (aliased)

    const int tid  = threadIdx.x;
    const int lane = tid & 31;
    const int wid  = tid >> 5;        // wave id 0..3  == output-d offset in tile
    const int hi   = lane >> 4;       // K-half (B) / M-half (C/D)
    const int col  = lane & 15;       // N column within a 16-wide tile

    int bid = blockIdx.x;
    int n   = bid / 432;
    int rem = bid % 432;
    int bd  = rem / 36;
    int r2  = rem % 36;
    int bh  = r2 / 3;
    int bw  = r2 % 3;
    int d0  = bd * 4, h0 = bh * 4, w0 = bw * 16;

    v8f acc[2][4];
#pragma unroll
    for (int mt = 0; mt < 2; ++mt)
#pragma unroll
        for (int nt = 0; nt < 4; ++nt) acc[mt][nt] = (v8f){};

    for (int cc = 0; cc < 4; ++cc) {
        __syncthreads();
        // ---- stage 8x8x20 x 32ch halo slab, f32 -> f16, channel-last ----
        int ci0 = cc * 32;
        for (int it = 0; it < (NPOS * 32) / 128; ++it) {   // 320 iters
            int idx = it * 128 + tid;
            int dx  = idx % 20;
            int t2  = idx / 20;
            int dy  = t2 & 7;
            int t3  = t2 >> 3;
            int dz  = t3 & 7;
            int ci  = t3 >> 3;
            int c = ci0 + ci;
            int d = d0 - 2 + dz, h = h0 - 2 + dy, w = w0 - 2 + dx;
            float v = 0.0f;
            if (c < CIN && (unsigned)d < 48u && (unsigned)h < 48u && (unsigned)w < 48u)
                v = x[(size_t)(n * CIN + c) * SPAT + d * 2304 + h * 48 + w];
            lds_x[((dz * 8 + dy) * 20 + dx) * POSH + ci] = (_Float16)v;
        }
        __syncthreads();

        // ---- 125 K-steps (one per kernel offset, K=32 channels each) ----
        const _Float16* Abase =
            Afrag + (size_t)cc * 125 * 1024 + (size_t)lane * 16;
#pragma unroll 1
        for (int ko = 0; ko < 125; ++ko) {
            int kd = ko / 25;
            int kr = ko % 25;
            int kh = kr / 5;
            int kw = kr % 5;
            const _Float16* ap = Abase + (size_t)ko * 1024;
            // one base address; the 4 N-tiles are +nt*800 halfs (DS imm offsets)
            const _Float16* bb =
                &lds_x[(((wid + kd) * 8 + kh) * 20 + kw + col) * POSH + 8 * hi];

            v16h b[4];
#pragma unroll
            for (int nt = 0; nt < 4; ++nt) {
                v8h blo = *(const v8h*)(bb + nt * 800);
                v8h bhi = *(const v8h*)(bb + nt * 800 + 16);
                b[nt] = __builtin_shufflevector(blo, bhi,
                        0,1,2,3,4,5,6,7,8,9,10,11,12,13,14,15);
            }
            v16h a0 = *(const v16h*)(ap);          // M-tile 0 (co 0..15)
            v16h a1 = *(const v16h*)(ap + 512);    // M-tile 1 (co 16..31)

            // 8 back-to-back WMMAs on independent accumulators: no hazards,
            // in-place C=D accumulation.
#pragma unroll
            for (int nt = 0; nt < 4; ++nt)
                acc[0][nt] = __builtin_amdgcn_wmma_f32_16x16x32_f16(
                    false, a0, false, b[nt], (short)0, acc[0][nt], false, false);
#pragma unroll
            for (int nt = 0; nt < 4; ++nt)
                acc[1][nt] = __builtin_amdgcn_wmma_f32_16x16x32_f16(
                    false, a1, false, b[nt], (short)0, acc[1][nt], false, false);
        }
    }

    // ---- sigmoid gates -> LDS (C/D layout: lane,VGPR r -> M=r+8*hi, N=col) ----
    // sigmoid(v) = rcp(1 + exp2(-v*log2e)) : native v_exp_f32 + v_rcp_f32.
    __syncthreads();
#pragma unroll
    for (int mt = 0; mt < 2; ++mt)
#pragma unroll
        for (int nt = 0; nt < 4; ++nt)
#pragma unroll
            for (int r = 0; r < 8; ++r) {
                int co = mt * 16 + r + 8 * hi;
                if (co < 24) {
                    float v = acc[mt][nt][r];
                    float e = __builtin_amdgcn_exp2f(v * -1.44269504088896340736f);
                    float g = __builtin_amdgcn_rcpf(1.0f + e);
                    lds_g[co * 256 + (wid * 4 + nt) * 16 + col] = g;
                }
            }
    __syncthreads();

    // ---- epilogue: relu scalars, gate l=1 (16x3) and l=2 (8x5) fields ----
    for (int i = tid; i < CIN * 256; i += 128) {
        int c  = i >> 8;
        int p  = i & 255;
        int px = p & 15, py = (p >> 4) & 3, pz = p >> 6;
        int d = d0 + pz, h = h0 + py, w = w0 + px;
        size_t gidx = (size_t)(n * CIN + c) * SPAT + d * 2304 + h * 48 + w;
        float xv = x[gidx];
        float o;
        if (c < 32) {
            o = fmaxf(xv, 0.0f);
        } else if (c < 80) {
            int m = (c - 32) / 3;               // gate 0..15
            o = xv * lds_g[m * 256 + p];
        } else {
            int m = 16 + (c - 80) / 5;          // gate 16..23
            o = xv * lds_g[m * 256 + p];
        }
        out[gidx] = o;
    }
}

extern "C" void kernel_launch(void* const* d_in, const int* in_sizes, int n_in,
                              void* d_out, int out_size, void* d_ws, size_t ws_size,
                              hipStream_t stream)
{
    const float* x  = (const float*)d_in[0];       // (2,120,48,48,48) f32
    const float* wg = (const float*)d_in[1];       // (24,120,5,5,5)   f32
    float* out = (float*)d_out;                    // (2,120,48,48,48) f32
    _Float16* A = (_Float16*)d_ws;                 // 1,024,000 B fragment-packed weights

    GatedAct_packA<<<125, 256, 0, stream>>>(wg, A);
    GatedAct_conv<<<864, 128, 0, stream>>>(x, A, out);
}
